// CDFG_31379031064610
// MI455X (gfx1250) — matrix operand order repaired and checked by reading, not compiled, not run on hardware
//
#include <hip/hip_runtime.h>
#include <hip/hip_bf16.h>
#include <math.h>

// ---------------- CDNA5 WMMA types ----------------
typedef __attribute__((ext_vector_type(16))) __bf16 v16bf;
typedef __attribute__((ext_vector_type(8)))  float  v8f;
typedef __attribute__((ext_vector_type(4)))  float  v4f;   // native vector (NT loads)

#define TB 256
#define BM 64
#define BN 128
#define BK 32

// round-to-nearest f32->bf16
__device__ __forceinline__ unsigned short f2bf(float f) {
  return (unsigned short)((__float_as_uint(f) + 0x8000u) >> 16);
}
// pack two f32 -> bf16x2 in one dword: 2x v_add + 1x v_perm_b32
__device__ __forceinline__ unsigned pack_bf16(float a, float b) {
  unsigned ua = __float_as_uint(a) + 0x8000u;
  unsigned ub = __float_as_uint(b) + 0x8000u;
  // bytes: [0]=ua.b2 [1]=ua.b3 [2]=ub.b2 [3]=ub.b3  (sel source: {ub:ua})
  return __builtin_amdgcn_perm(ub, ua, 0x07060302u);
}
__device__ __forceinline__ float bf2f(unsigned short b) {
  return __uint_as_float(((unsigned int)b) << 16);
}

union BfFrag { unsigned int u[8]; v16bf v; };

// one 16-byte async global->LDS copy (per active lane)
__device__ __forceinline__ void async_b128(unsigned lds_off, const void* src) {
  asm volatile("global_load_async_to_lds_b128 %0, %1, off"
               :: "v"(lds_off), "v"(src) : "memory");
}
__device__ __forceinline__ void wait_async0() {
  asm volatile("s_wait_asynccnt 0x0" ::: "memory");
}

// ---------------------------------------------------------------------------
// fp32-input GEMM: C = act(A[M,K] @ B + bias). bf16 WMMA, f32 accumulate.
// Double-buffered LDS; next tile's global loads overlap current WMMAs.
// transB : B(k,n) = B[n*K+k].  maskA: A -> (A>0.5 ? A : 0).
// act: 0 none, 1 relu, 2 elu.
// outMode: 0 -> fp32 C[m*N+n]; 1 -> bf16 transposed CT[n*M+m] (no bias/act)
// Requires: M%64==0, K%32==0, N%16==0.
// ---------------------------------------------------------------------------
__global__ __launch_bounds__(TB) void gemm_f32in(
    const float* __restrict__ A, const float* __restrict__ B,
    const float* __restrict__ bias, float* __restrict__ C,
    unsigned short* __restrict__ CT,
    int M, int N, int K, int transB, int act, int maskA, int outMode)
{
  __shared__ unsigned short sA[2][BM * BK];
  __shared__ unsigned short sB[2][BN * BK];

  const int tid  = threadIdx.x;
  const int lane = tid & 31;
  const int wid  = tid >> 5;
  const int wm   = (wid >> 2) * 32;
  const int wn   = (wid & 3) * 32;
  const int mblk = blockIdx.y * BM;
  const int nblk = blockIdx.x * BN;
  const int half = lane >> 4;
  const int lm   = lane & 15;

  v8f acc[2][2] = {};
  float4 ra[2], rb[4];

  auto loadTiles = [&](int k0) {
    #pragma unroll
    for (int it = 0; it < 2; ++it) {
      int c = tid + it * TB;              // 512 float4 chunks of A
      int row = c >> 3, kq = (c & 7) * 4;
      ra[it] = *(const float4*)(A + (size_t)(mblk + row) * K + (k0 + kq));
    }
    if (transB) {
      #pragma unroll
      for (int it = 0; it < 4; ++it) {
        int c = tid + it * TB;            // k-contiguous rows of B^T
        int n = c >> 3, kq = (c & 7) * 4;
        int gn = nblk + n; if (gn > N - 1) gn = N - 1;   // cols >= N discarded
        rb[it] = *(const float4*)(B + (size_t)gn * K + (k0 + kq));
      }
    } else {
      #pragma unroll
      for (int it = 0; it < 4; ++it) {
        int c = tid + it * TB;            // n-contiguous rows of B
        int k = c >> 5, nq = (c & 31) * 4;
        int gn = nblk + nq; if (gn > N - 4) gn = N - 4;  // clamp (N%16==0)
        rb[it] = *(const float4*)(B + (size_t)(k0 + k) * N + gn);
      }
    }
  };
  auto storeTiles = [&](int buf) {
    #pragma unroll
    for (int it = 0; it < 2; ++it) {
      int c = tid + it * TB;
      int row = c >> 3, kq = (c & 7) * 4;
      float4 v4 = ra[it];
      if (maskA) {
        v4.x = (v4.x > 0.5f) ? v4.x : 0.0f;
        v4.y = (v4.y > 0.5f) ? v4.y : 0.0f;
        v4.z = (v4.z > 0.5f) ? v4.z : 0.0f;
        v4.w = (v4.w > 0.5f) ? v4.w : 0.0f;
      }
      uint2 p; p.x = pack_bf16(v4.x, v4.y); p.y = pack_bf16(v4.z, v4.w);
      *(uint2*)&sA[buf][row * BK + kq] = p;
    }
    if (transB) {
      #pragma unroll
      for (int it = 0; it < 4; ++it) {
        int c = tid + it * TB;
        int n = c >> 3, kq = (c & 7) * 4;
        uint2 p; p.x = pack_bf16(rb[it].x, rb[it].y); p.y = pack_bf16(rb[it].z, rb[it].w);
        *(uint2*)&sB[buf][n * BK + kq] = p;
      }
    } else {
      #pragma unroll
      for (int it = 0; it < 4; ++it) {
        int c = tid + it * TB;
        int k = c >> 5, nq = (c & 31) * 4;
        sB[buf][(nq + 0) * BK + k] = f2bf(rb[it].x);
        sB[buf][(nq + 1) * BK + k] = f2bf(rb[it].y);
        sB[buf][(nq + 2) * BK + k] = f2bf(rb[it].z);
        sB[buf][(nq + 3) * BK + k] = f2bf(rb[it].w);
      }
    }
  };
  auto compute = [&](int buf) {
    const unsigned int* pa0 = (const unsigned int*)sA[buf];
    const unsigned int* pb0 = (const unsigned int*)sB[buf];
    #pragma unroll
    for (int i = 0; i < 2; ++i) {
      BfFrag af;
      const unsigned int* pa = pa0 + (size_t)(wm + i * 16 + lm) * (BK / 2);
      #pragma unroll
      for (int j = 0; j < 4; ++j) {
        af.u[j]     = pa[half * 8 + j];
        af.u[4 + j] = pa[half * 8 + 4 + j];
      }
      #pragma unroll
      for (int j = 0; j < 2; ++j) {
        BfFrag bf;
        const unsigned int* pb = pb0 + (size_t)(wn + j * 16 + lm) * (BK / 2);
        #pragma unroll
        for (int r = 0; r < 8; ++r) bf.u[r] = pb[half * 8 + r];
        acc[i][j] = __builtin_amdgcn_wmma_f32_16x16x32_bf16(
            false, af.v, false, bf.v, (short)0, acc[i][j], false, false);
      }
    }
  };

  const int nk = K / BK;
  loadTiles(0);
  for (int kt = 0; kt < nk; ++kt) {
    const int cur = kt & 1;
    storeTiles(cur);
    __syncthreads();
    if (kt + 1 < nk) loadTiles((kt + 1) * BK);  // overlaps the WMMAs below
    compute(cur);
    // next iteration stores into buffer cur^1, last read before the barrier
  }

  #pragma unroll
  for (int i = 0; i < 2; ++i) {
    #pragma unroll
    for (int j = 0; j < 2; ++j) {
      int ng = nblk + wn + j * 16 + lm;
      if (ng >= N) continue;
      int mg0 = mblk + wm + i * 16 + half * 8;
      if (outMode == 0) {
        float bb = bias ? bias[ng] : 0.0f;
        #pragma unroll
        for (int r = 0; r < 8; ++r) {
          float v = acc[i][j][r] + bb;
          if (act == 1)      v = fmaxf(v, 0.0f);
          else if (act == 2) v = (v > 0.0f) ? v : (__expf(v) - 1.0f);
          C[(size_t)(mg0 + r) * N + ng] = v;
        }
      } else {
        uint4 p;
        p.x = pack_bf16(acc[i][j][0], acc[i][j][1]);
        p.y = pack_bf16(acc[i][j][2], acc[i][j][3]);
        p.z = pack_bf16(acc[i][j][4], acc[i][j][5]);
        p.w = pack_bf16(acc[i][j][6], acc[i][j][7]);
        *(uint4*)(CT + (size_t)ng * M + mg0) = p;
      }
    }
  }
}

// ---------------------------------------------------------------------------
// bf16-input GEMM: C = act(A @ B). A: bf16 [M,K]. B: bmode==0 -> bf16 [N,K]
// (async global->LDS b128 copies, double-buffered so copies for tile k+1 are
// in flight during tile k's WMMAs); bmode==1 -> fp32 [K,N] (small N).
// ---------------------------------------------------------------------------
__global__ __launch_bounds__(TB) void gemm_bf16in(
    const unsigned short* __restrict__ A, const unsigned short* __restrict__ Bt,
    const float* __restrict__ Bf, float* __restrict__ C,
    int M, int N, int K, int bmode, int act)
{
  __shared__ unsigned short sA[2][BM * BK];
  __shared__ unsigned short sB[2][BN * BK];

  const int tid  = threadIdx.x;
  const int lane = tid & 31;
  const int wid  = tid >> 5;
  const int wm   = (wid >> 2) * 32;
  const int wn   = (wid & 3) * 32;
  const int mblk = blockIdx.y * BM;
  const int nblk = blockIdx.x * BN;
  const int half = lane >> 4;
  const int lm   = lane & 15;

  v8f acc[2][2] = {};

  auto stage = [&](int buf, int k0) {
    { // A tile: 4KB pure byte copy, one async b128 per lane
      int row = tid >> 2, k8 = (tid & 3) * 8;
      unsigned lds = (unsigned)(uintptr_t)&sA[buf][row * BK + k8];
      async_b128(lds, (const void*)(A + (size_t)(mblk + row) * K + (k0 + k8)));
    }
    if (bmode == 0) {  // B tile: 8KB async copy from bf16 [N,K]
      #pragma unroll
      for (int it = 0; it < 2; ++it) {
        int c = tid + it * TB;
        int n = c >> 2, k8 = (c & 3) * 8;
        int gn = nblk + n; if (gn > N - 1) gn = N - 1;   // cols >= N discarded
        unsigned lds = (unsigned)(uintptr_t)&sB[buf][n * BK + k8];
        async_b128(lds, (const void*)(Bt + (size_t)gn * K + (k0 + k8)));
      }
    } else {           // B from fp32 [K,N]
      for (int i = tid; i < BN * BK; i += TB) {
        int n = i >> 5, k = i & 31;
        int gn = nblk + n; if (gn > N - 1) gn = N - 1;
        sB[buf][n * BK + k] = f2bf(Bf[(size_t)(k0 + k) * N + gn]);
      }
    }
  };
  auto compute = [&](int buf) {
    const unsigned int* pa0 = (const unsigned int*)sA[buf];
    const unsigned int* pb0 = (const unsigned int*)sB[buf];
    #pragma unroll
    for (int i = 0; i < 2; ++i) {
      BfFrag af;
      const unsigned int* pa = pa0 + (size_t)(wm + i * 16 + lm) * (BK / 2);
      #pragma unroll
      for (int j = 0; j < 4; ++j) {
        af.u[j]     = pa[half * 8 + j];
        af.u[4 + j] = pa[half * 8 + 4 + j];
      }
      #pragma unroll
      for (int j = 0; j < 2; ++j) {
        BfFrag bf;
        const unsigned int* pb = pb0 + (size_t)(wn + j * 16 + lm) * (BK / 2);
        #pragma unroll
        for (int r = 0; r < 8; ++r) bf.u[r] = pb[half * 8 + r];
        acc[i][j] = __builtin_amdgcn_wmma_f32_16x16x32_bf16(
            false, af.v, false, bf.v, (short)0, acc[i][j], false, false);
      }
    }
  };

  const int nk = K / BK;
  stage(0, 0);
  for (int kt = 0; kt < nk; ++kt) {
    const int cur = kt & 1;
    wait_async0();            // this wave's copies for 'cur' are complete
    __syncthreads();          // all waves' copies complete
    if (kt + 1 < nk) stage(cur ^ 1, (kt + 1) * BK);  // flies during WMMAs
    compute(cur);
    // buffer cur^1 was last read before this iteration's barrier -> safe
  }

  #pragma unroll
  for (int i = 0; i < 2; ++i) {
    #pragma unroll
    for (int j = 0; j < 2; ++j) {
      int ng = nblk + wn + j * 16 + lm;
      if (ng >= N) continue;
      int mg0 = mblk + wm + i * 16 + half * 8;
      #pragma unroll
      for (int r = 0; r < 8; ++r) {
        float v = acc[i][j][r];
        if (act == 1)      v = fmaxf(v, 0.0f);
        else if (act == 2) v = (v > 0.0f) ? v : (__expf(v) - 1.0f);
        C[(size_t)(mg0 + r) * N + ng] = v;
      }
    }
  }
}

// f_self/f_neigh from bf16 transposed H: fs[m] = sum_d Ht[d*Mn+m]*a_s[d]
__global__ __launch_bounds__(TB) void dual_dot_T(
    const unsigned short* __restrict__ Ht, const float* __restrict__ a_s,
    const float* __restrict__ a_n, float* __restrict__ fs,
    float* __restrict__ fn, int D, int Mn)
{
  const int node = blockIdx.x * TB + threadIdx.x;
  float s1 = 0.0f, s2 = 0.0f;
  for (int d = 0; d < D; ++d) {
    float h = bf2f(Ht[(size_t)d * Mn + node]);
    s1 += h * a_s[d];
    s2 += h * a_n[d];
  }
  fs[node] = s1;
  fn[node] = s2;
}

// E[i,j] = adj>0.5 ? leaky((fs_i+fn_j)*M_ij, 0.2) : -9e15
// vectorized; adj/M streamed with non-temporal hints (keep L2 for the bf16
// attention/h working set the WMMA GEMMs re-read). NT builtin needs native
// clang ext-vector types, not HIP_vector_type.
__global__ void gat_edge(const float* __restrict__ adj,
                         const float* __restrict__ Mm,
                         const float* __restrict__ fs,
                         const float* __restrict__ fn,
                         float* __restrict__ E)
{
  const size_t total4 = (size_t)4096 * 1024;   // N*N/4
  const v4f* adj4 = (const v4f*)adj;
  const v4f* Mm4  = (const v4f*)Mm;
  const v4f* fn4  = (const v4f*)fn;
  v4f* E4 = (v4f*)E;
  for (size_t idx = blockIdx.x * (size_t)blockDim.x + threadIdx.x; idx < total4;
       idx += (size_t)gridDim.x * blockDim.x) {
    int i  = (int)(idx >> 10);
    int j4 = (int)(idx & 1023);
    float fi = fs[i];
    v4f fj = fn4[j4];
    v4f m4 = __builtin_nontemporal_load(&Mm4[idx]);
    v4f a4 = __builtin_nontemporal_load(&adj4[idx]);
    v4f r;
    float e;
    e = (fi + fj.x) * m4.x; e = (e > 0.f) ? e : 0.2f * e; r.x = (a4.x > 0.5f) ? e : -9.0e15f;
    e = (fi + fj.y) * m4.y; e = (e > 0.f) ? e : 0.2f * e; r.y = (a4.y > 0.5f) ? e : -9.0e15f;
    e = (fi + fj.z) * m4.z; e = (e > 0.f) ? e : 0.2f * e; r.z = (a4.z > 0.5f) ? e : -9.0e15f;
    e = (fi + fj.w) * m4.w; e = (e > 0.f) ? e : 0.2f * e; r.w = (a4.w > 0.5f) ? e : -9.0e15f;
    E4[idx] = r;
  }
}

// row softmax fp32 [rows,C] -> bf16 out (C even; paired b32 stores)
__global__ __launch_bounds__(TB) void row_softmax_bf16(
    const float* __restrict__ X, unsigned short* __restrict__ Y, int C)
{
  const int row = blockIdx.x, tid = threadIdx.x;
  const float* x = X + (size_t)row * C;
  unsigned* y2 = (unsigned*)(Y + (size_t)row * C);
  __shared__ float red[TB];
  float mx = -INFINITY;
  for (int j = tid; j < C; j += TB) mx = fmaxf(mx, x[j]);
  red[tid] = mx; __syncthreads();
  for (int s = TB / 2; s > 0; s >>= 1) {
    if (tid < s) red[tid] = fmaxf(red[tid], red[tid + s]);
    __syncthreads();
  }
  mx = red[0]; __syncthreads();
  float sum = 0.0f;
  for (int j = tid; j < C; j += TB) sum += __expf(x[j] - mx);
  red[tid] = sum; __syncthreads();
  for (int s = TB / 2; s > 0; s >>= 1) {
    if (tid < s) red[tid] += red[tid + s];
    __syncthreads();
  }
  float inv = 1.0f / red[0];
  const int C2 = C >> 1;
  for (int j = tid; j < C2; j += TB) {
    float2 v = *(const float2*)(x + 2 * j);
    y2[j] = pack_bf16(__expf(v.x - mx) * inv, __expf(v.y - mx) * inv);
  }
}

// in-place fp32 row softmax (predict, C=16)
__global__ __launch_bounds__(TB) void row_softmax(float* __restrict__ X, int C)
{
  const int row = blockIdx.x, tid = threadIdx.x;
  float* x = X + (size_t)row * C;
  __shared__ float red[TB];
  float mx = -INFINITY;
  for (int j = tid; j < C; j += TB) mx = fmaxf(mx, x[j]);
  red[tid] = mx; __syncthreads();
  for (int s = TB / 2; s > 0; s >>= 1) {
    if (tid < s) red[tid] = fmaxf(red[tid], red[tid + s]);
    __syncthreads();
  }
  mx = red[0]; __syncthreads();
  float sum = 0.0f;
  for (int j = tid; j < C; j += TB) sum += __expf(x[j] - mx);
  red[tid] = sum; __syncthreads();
  for (int s = TB / 2; s > 0; s >>= 1) {
    if (tid < s) red[tid] += red[tid + s];
    __syncthreads();
  }
  float inv = 1.0f / red[0];
  for (int j = tid; j < C; j += TB) x[j] = __expf(x[j] - mx) * inv;
}

__global__ void blend_k(float* __restrict__ out, const float* __restrict__ a,
                        const float* __restrict__ b, size_t n)
{
  for (size_t i = blockIdx.x * (size_t)blockDim.x + threadIdx.x; i < n;
       i += (size_t)gridDim.x * blockDim.x)
    out[i] = 0.9f * a[i] + 0.1f * b[i];
}

// Student-t, V=1: q = (1+d2)^-1 row-normalized. NZ=64, NC=16.
__global__ __launch_bounds__(64) void student_q(
    const float* __restrict__ Z, const float* __restrict__ cl,
    float* __restrict__ q)
{
  const int i = blockIdx.x, tid = threadIdx.x;
  __shared__ float zrow[64];
  __shared__ float qs[16];
  zrow[tid] = Z[(size_t)i * 64 + tid];
  __syncthreads();
  if (tid < 16) {
    float d2 = 0.0f;
    #pragma unroll
    for (int d = 0; d < 64; ++d) {
      float t = zrow[d] - cl[tid * 64 + d];
      d2 += t * t;
    }
    qs[tid] = 1.0f / (1.0f + d2);
  }
  __syncthreads();
  if (tid == 0) {
    float s = 0.0f;
    for (int c = 0; c < 16; ++c) s += qs[c];
    float inv = 1.0f / s;
    for (int c = 0; c < 16; ++c) q[(size_t)i * 16 + c] = qs[c] * inv;
  }
}

// ---------------------------------------------------------------------------
extern "C" void kernel_launch(void* const* d_in, const int* in_sizes, int n_in,
                              void* d_out, int out_size, void* d_ws, size_t ws_size,
                              hipStream_t stream)
{
  (void)in_sizes; (void)n_in; (void)out_size; (void)ws_size;
  const int N = 4096;
  const float* x   = (const float*)d_in[0];
  const float* adj = (const float*)d_in[1];
  const float* Mm  = (const float*)d_in[2];
  auto P = [&](int i) { return (const float*)d_in[i]; };

  float* out    = (float*)d_out;
  float* xbar_o = out;
  float* q_o    = xbar_o + (size_t)N * 1024;
  float* pred_o = q_o + (size_t)N * 16;
  float* z_o    = pred_o + (size_t)N * 16;
  float* h4_o   = z_o + (size_t)N * 64;

  char* cur = (char*)d_ws;
  auto allocB = [&](size_t bytes) {
    void* p = cur; cur += (bytes + 255) & ~(size_t)255; return p;
  };
  float* t1   = (float*)allocB((size_t)N * 512 * 4);
  float* t2   = (float*)allocB((size_t)N * 512 * 4);
  float* t3   = (float*)allocB((size_t)N * 2048 * 4);
  float* d1   = (float*)allocB((size_t)N * 2048 * 4);  // dec1; reused for blends
  float* d2   = (float*)allocB((size_t)N * 512 * 4);
  float* d3   = (float*)allocB((size_t)N * 512 * 4);
  float* h1   = (float*)allocB((size_t)N * 512 * 4);
  float* h2   = (float*)allocB((size_t)N * 512 * 4);
  float* h3   = (float*)allocB((size_t)N * 2048 * 4);
  float* E    = (float*)allocB((size_t)N * N * 4);          // edge scores / G
  unsigned short* attB  = (unsigned short*)allocB((size_t)N * N * 2);    // bf16 att / s
  unsigned short* hbufT = (unsigned short*)allocB((size_t)2048 * N * 2); // bf16 h^T
  float* fs   = (float*)allocB((size_t)N * 4);
  float* fn   = (float*)allocB((size_t)N * 4);
  float* zb   = (float*)allocB((size_t)N * 64 * 4);
  float* z5   = (float*)allocB((size_t)N * 16 * 4);
  float* hg   = (float*)allocB((size_t)N * 16 * 4);

  auto gemmF = [&](const float* A, const float* B, const float* bias, float* C,
                   unsigned short* CT, int Mr, int Nc, int Kd,
                   int transB, int act, int maskA, int outMode) {
    dim3 grid((Nc + BN - 1) / BN, Mr / BM);
    gemm_f32in<<<grid, TB, 0, stream>>>(A, B, bias, C, CT, Mr, Nc, Kd,
                                        transB, act, maskA, outMode);
  };
  auto gemmB = [&](const unsigned short* A, const unsigned short* Bt,
                   const float* Bf, float* C, int Mr, int Nc, int Kd,
                   int bmode, int act) {
    dim3 grid((Nc + BN - 1) / BN, Mr / BM);
    gemm_bf16in<<<grid, TB, 0, stream>>>(A, Bt, Bf, C, Mr, Nc, Kd, bmode, act);
  };

  // ---------------- AE ----------------
  gemmF(x,   P(3),  P(4),  t1,     nullptr, N, 512,  1024, 0, 1, 0, 0);
  gemmF(t1,  P(5),  P(6),  t2,     nullptr, N, 512,  512,  0, 1, 0, 0);
  gemmF(t2,  P(7),  P(8),  t3,     nullptr, N, 2048, 512,  0, 1, 0, 0);
  gemmF(t3,  P(9),  P(10), z_o,    nullptr, N, 64,   2048, 0, 0, 0, 0);
  gemmF(z_o, P(11), P(12), d1,     nullptr, N, 2048, 64,   0, 1, 0, 0);
  gemmF(d1,  P(13), P(14), d2,     nullptr, N, 512,  2048, 0, 1, 0, 0);
  gemmF(d2,  P(15), P(16), d3,     nullptr, N, 512,  512,  0, 1, 0, 0);
  gemmF(d3,  P(17), P(18), xbar_o, nullptr, N, 1024, 512,  0, 0, 0, 0);

  // ---------------- GAT stack ----------------
  auto gat = [&](const float* X, int din, int dout, int wi, float* hout) {
    gemmF(X, P(wi), nullptr, nullptr, hbufT, N, dout, din, 0, 0, 0, 1);
    dual_dot_T<<<N / TB, TB, 0, stream>>>(hbufT, P(wi + 1), P(wi + 2),
                                          fs, fn, dout, N);
    gat_edge<<<8192, TB, 0, stream>>>(adj, Mm, fs, fn, E);
    row_softmax_bf16<<<N, TB, 0, stream>>>(E, attB, N);
    gemmB(attB, hbufT, nullptr, hout, N, dout, N, 0, 2);  // elu(att@h)
  };

  gat(x, 1024, 512, 19, h1);
  blend_k<<<4096, TB, 0, stream>>>(d1, h1, t1, (size_t)N * 512);
  gat(d1, 512, 512, 22, h2);
  blend_k<<<4096, TB, 0, stream>>>(d1, h2, t2, (size_t)N * 512);
  gat(d1, 512, 2048, 25, h3);
  blend_k<<<4096, TB, 0, stream>>>(d1, h3, t3, (size_t)N * 2048);
  gat(d1, 2048, 64, 28, h4_o);

  // ---------------- GNN layer: adj_mat @ ((0.9*h4 + 0.1*z) @ g5_w) ----------
  blend_k<<<4096, TB, 0, stream>>>(zb, h4_o, z_o, (size_t)N * 64);
  gemmF(zb,  P(31), nullptr, z5, nullptr, N, 16, 64, 0, 0, 0, 0);
  gemmF(adj, z5,    nullptr, hg, nullptr, N, 16, N,  0, 0, 1, 0);

  // ---------------- s = softmax(h1@h1^T); z3 = s@hg ----------
  gemmF(h1, h1, nullptr, E, nullptr, N, N, 512, 1, 0, 0, 0);
  row_softmax_bf16<<<N, TB, 0, stream>>>(E, attB, N);
  gemmB(attB, nullptr, hg, pred_o, N, 16, N, 1, 0);
  row_softmax<<<N, TB, 0, stream>>>(pred_o, 16);

  // ---------------- Student-t q ----------------
  student_q<<<N, 64, 0, stream>>>(z_o, P(32), q_o);
}
